// TMDConv_30588757083009
// MI455X (gfx1250) — matrix operand order repaired and checked by reading, not compiled, not run on hardware
//
#include <hip/hip_runtime.h>
#include <hip/hip_bf16.h>
#include <math.h>

#define FDIM 128
#define F3   (3 * FDIM)
#define LRBF 20
#define RCUT 5.0f
#define PI_F 3.14159265358979f
#define THREADS 256

typedef __attribute__((ext_vector_type(16))) __bf16 v16bf;
typedef __attribute__((ext_vector_type(8)))  float  v8f;
typedef __attribute__((ext_vector_type(4)))  float  f32x4;
typedef __attribute__((ext_vector_type(8)))  unsigned short u16x8;
typedef __attribute__((ext_vector_type(4)))  unsigned short u16x4;

union Frag { v16bf v; u16x8 h[2]; };

__device__ __forceinline__ unsigned short f2bf(float f) {
  unsigned int u = __float_as_uint(f);
  u += 0x7FFFu + ((u >> 16) & 1u);   // round-to-nearest-even
  return (unsigned short)(u >> 16);
}

__device__ __forceinline__ float ssp(float x) {
  // softplus(x) - log(2), via hardware exp/log transcendentals
  return fmaxf(x, 0.0f) + __logf(1.0f + __expf(-fabsf(x))) - 0.69314718056f;
}

// A fragment: 16x32 bf16, lane(0..15)=row M, K chunks at k0+h*8 and k0+16+h*8
__device__ __forceinline__ v16bf load_fragA(const unsigned short* base, int pitch,
                                            int k0, int nl, int hh) {
  Frag f;
  const unsigned short* p = base + nl * pitch + k0 + hh * 8;
  f.h[0] = *reinterpret_cast<const u16x8*>(p);
  f.h[1] = *reinterpret_cast<const u16x8*>(p + 16);
  return f.v;
}

// B fragment: 32x16 bf16, lane(0..15)=col N, K chunks at k0+h*8 and k0+16+h*8
__device__ __forceinline__ v16bf load_fragB(const unsigned short* base, int pitch,
                                            int k0, int nl, int hh) {
  Frag f;
  const unsigned short* p = base + nl * pitch + k0 + hh * 8;
  f.h[0] = *reinterpret_cast<const u16x8*>(p);
  f.h[1] = *reinterpret_cast<const u16x8*>(p + 16);
  return f.v;
}

// gather one row-major fp32 row segment as 4 bf16 into LDS (8B store)
__device__ __forceinline__ void gather4_bf16(unsigned short* dstLds,
                                             const float* __restrict__ srcRow) {
  f32x4 x = *reinterpret_cast<const f32x4*>(srcRow);
  u16x4 o;
  o.x = f2bf(x.x); o.y = f2bf(x.y); o.z = f2bf(x.z); o.w = f2bf(x.w);
  *reinterpret_cast<u16x4*>(dstLds) = o;
}

// ---------------------------------------------------------------- init
__global__ void init_kernel(const float* __restrict__ v, const float* __restrict__ s,
                            float* v_new, float* s_new, float* sum_s2, float* uv_sum,
                            float* deg, int N) {
  long long NF  = (long long)N * FDIM;
  long long NF3 = NF * 3;   // == N*F3 as well
  long long stride = (long long)gridDim.x * blockDim.x;
  for (long long i = (long long)blockIdx.x * blockDim.x + threadIdx.x; i < NF3; i += stride) {
    v_new[i]  = v[i];
    uv_sum[i] = 0.0f;
    sum_s2[i] = 0.0f;
    if (i < NF) s_new[i] = s[i];
    if (i < N)  deg[i] = 0.0f;
  }
}

// ------------------------------------------------------- weight -> bf16
__global__ void cvt_kernel(const float* __restrict__ w1, const float* __restrict__ w2,
                           const float* __restrict__ u1, const float* __restrict__ u2,
                           unsigned short* out) {
  const int n1 = FDIM * FDIM, n2 = F3 * FDIM;
  const int total = 2 * n1 + 2 * n2;
  for (int i = blockIdx.x * blockDim.x + threadIdx.x; i < total;
       i += gridDim.x * blockDim.x) {
    float x;
    if (i < n1)                x = w1[i];
    else if (i < n1 + n2)      x = w2[i - n1];
    else if (i < 2 * n1 + n2)  x = u1[i - n1 - n2];
    else                       x = u2[i - 2 * n1 - n2];
    out[i] = f2bf(x);
  }
}

// ---------------------------------------------------------------- degree
__global__ void deg_kernel(const int* __restrict__ dst, float* deg, int E) {
  int e = blockIdx.x * blockDim.x + threadIdx.x;
  if (e < E) atomicAdd(&deg[dst[e]], 1.0f);
}

// ---------------------------------------------------------------- pass 1
__global__ void __launch_bounds__(THREADS)
pass1_kernel(const float* __restrict__ x, const float* __restrict__ v,
             const float* __restrict__ s,
             const unsigned short* __restrict__ w1, const float* __restrict__ b1,
             const unsigned short* __restrict__ w2, const float* __restrict__ b2,
             const float* __restrict__ mv_w, const float* __restrict__ mv_b,
             const int* __restrict__ src, const int* __restrict__ dst,
             float* s_new, float* v_new, int E) {
  __shared__ unsigned short sA[16 * FDIM];   // gathered s[src], bf16
  __shared__ unsigned short sH[16 * FDIM];   // ssp(hidden), bf16
  __shared__ float sPW[16 * F3];             // phi * w
  __shared__ float sRBF[16 * LRBF];
  __shared__ float sVec[16 * 4];             // unit vec xyz, r
  __shared__ int   sSrc[16], sDst[16];

  const int e0  = blockIdx.x * 16;
  const int tid = threadIdx.x;

  if (tid < 16) {
    int e  = e0 + tid;
    int sj = 0, dj = 0;
    if (e < E) { sj = src[e]; dj = dst[e]; }
    sSrc[tid] = sj; sDst[tid] = dj;
    float vx = x[sj * 3 + 0] - x[dj * 3 + 0];
    float vy = x[sj * 3 + 1] - x[dj * 3 + 1];
    float vz = x[sj * 3 + 2] - x[dj * 3 + 2];
    float r  = sqrtf(vx * vx + vy * vy + vz * vz + 1e-5f);
    float ir = 1.0f / r;
    sVec[tid * 4 + 0] = vx * ir; sVec[tid * 4 + 1] = vy * ir;
    sVec[tid * 4 + 2] = vz * ir; sVec[tid * 4 + 3] = r;
    float c = sqrtf(2.0f / RCUT) * ir;
#pragma unroll
    for (int l = 0; l < LRBF; l++)
      sRBF[tid * LRBF + l] = c * __sinf((float)(l + 1) * PI_F * r / RCUT);
  }
  __syncthreads();

  // gather s[src] -> LDS bf16 (float4 loads, 4x bf16 packed stores)
  for (int i = tid; i < 16 * (FDIM / 4); i += THREADS) {
    int m = i >> 5, f4 = (i & 31) * 4;
    gather4_bf16(sA + m * FDIM + f4, s + (long long)sSrc[m] * FDIM + f4);
  }
  __syncthreads();

  const int wave = tid >> 5, lane = tid & 31;
  const int nl = lane & 15, hh = lane >> 4;

  // GEMM1: h = ssp(S @ W1^T + b1), wave handles one 16-col tile
  {
    v8f acc = {};
    const unsigned short* bbase = w1 + wave * 16 * FDIM;
#pragma unroll
    for (int k0 = 0; k0 < FDIM; k0 += 32) {
      v16bf a = load_fragA(sA, FDIM, k0, nl, hh);
      v16bf b = load_fragB(bbase, FDIM, k0, nl, hh);
      acc = __builtin_amdgcn_wmma_f32_16x16x32_bf16(false, a, false, b,
                                                    (short)0, acc, false, false);
    }
    int n = wave * 16 + nl;
    float bias = b1[n];
#pragma unroll
    for (int r = 0; r < 8; r++) {
      int m = r + hh * 8;
      sH[m * FDIM + n] = f2bf(ssp(acc[r] + bias));
    }
  }
  __syncthreads();

  // GEMM2: phi = H @ W2^T + b2 ; then * cutoff(rbf @ mv_w^T + mv_b)
#pragma unroll
  for (int t = 0; t < 3; t++) {
    int ntile = wave + t * 8;   // 0..23
    v8f acc = {};
    const unsigned short* bbase = w2 + ntile * 16 * FDIM;
#pragma unroll
    for (int k0 = 0; k0 < FDIM; k0 += 32) {
      v16bf a = load_fragA(sH, FDIM, k0, nl, hh);
      v16bf b = load_fragB(bbase, FDIM, k0, nl, hh);
      acc = __builtin_amdgcn_wmma_f32_16x16x32_bf16(false, a, false, b,
                                                    (short)0, acc, false, false);
    }
    int n = ntile * 16 + nl;
    float bias2 = b2[n];
    float wb = mv_b[n];
    float mvw[LRBF];
#pragma unroll
    for (int l = 0; l < LRBF; l++) mvw[l] = mv_w[n * LRBF + l];
#pragma unroll
    for (int r = 0; r < 8; r++) {
      int m = r + hh * 8;
      float wl = wb;
#pragma unroll
      for (int l = 0; l < LRBF; l++) wl += sRBF[m * LRBF + l] * mvw[l];
      float wcut = (wl < RCUT) ? 0.5f * (__cosf(PI_F * wl / RCUT) + 1.0f) : 0.0f;
      sPW[m * F3 + n] = (acc[r] + bias2) * wcut;
    }
  }
  __syncthreads();

  // scatter: dv_e = vj*v_ + r_*(vec/r) ; ds_e = s_
  for (int i = tid; i < 16 * FDIM; i += THREADS) {
    int m = i >> 7, f = i & (FDIM - 1);
    int e = e0 + m;
    if (e >= E) continue;
    float v_ = sPW[m * F3 + f];
    float s_ = sPW[m * F3 + FDIM + f];
    float r_ = sPW[m * F3 + 2 * FDIM + f];
    int sj = sSrc[m], dj = sDst[m];
    float hx = sVec[m * 4 + 0];
    float hy = sVec[m * 4 + 1];
    float hz = sVec[m * 4 + 2];
    long long vb = ((long long)sj * FDIM + f) * 3;
    long long db = ((long long)dj * FDIM + f) * 3;
    atomicAdd(&v_new[db + 0], v[vb + 0] * v_ + r_ * hx);
    atomicAdd(&v_new[db + 1], v[vb + 1] * v_ + r_ * hy);
    atomicAdd(&v_new[db + 2], v[vb + 2] * v_ + r_ * hz);
    atomicAdd(&s_new[(long long)dj * FDIM + f], s_);
  }
}

// ---------------------------------------------------------------- pass 2
__global__ void __launch_bounds__(THREADS)
pass2_kernel(const unsigned short* __restrict__ w1, const float* __restrict__ b1,
             const unsigned short* __restrict__ w2, const float* __restrict__ b2,
             const int* __restrict__ src, const int* __restrict__ dst,
             const float* __restrict__ s_new, const float* __restrict__ v_new,
             float* sum_s2, float* uv_sum, int E) {
  __shared__ unsigned short sA[16 * FDIM];
  __shared__ unsigned short sH[16 * FDIM];
  __shared__ float sPW[16 * F3];
  __shared__ int sSrc[16], sDst[16];

  const int e0  = blockIdx.x * 16;
  const int tid = threadIdx.x;

  if (tid < 16) {
    int e = e0 + tid;
    sSrc[tid] = (e < E) ? src[e] : 0;
    sDst[tid] = (e < E) ? dst[e] : 0;
  }
  __syncthreads();

  for (int i = tid; i < 16 * (FDIM / 4); i += THREADS) {
    int m = i >> 5, f4 = (i & 31) * 4;
    gather4_bf16(sA + m * FDIM + f4, s_new + (long long)sSrc[m] * FDIM + f4);
  }
  __syncthreads();

  const int wave = tid >> 5, lane = tid & 31;
  const int nl = lane & 15, hh = lane >> 4;

  {
    v8f acc = {};
    const unsigned short* bbase = w1 + wave * 16 * FDIM;
#pragma unroll
    for (int k0 = 0; k0 < FDIM; k0 += 32) {
      v16bf a = load_fragA(sA, FDIM, k0, nl, hh);
      v16bf b = load_fragB(bbase, FDIM, k0, nl, hh);
      acc = __builtin_amdgcn_wmma_f32_16x16x32_bf16(false, a, false, b,
                                                    (short)0, acc, false, false);
    }
    int n = wave * 16 + nl;
    float bias = b1[n];
#pragma unroll
    for (int r = 0; r < 8; r++) {
      int m = r + hh * 8;
      sH[m * FDIM + n] = f2bf(ssp(acc[r] + bias));
    }
  }
  __syncthreads();

#pragma unroll
  for (int t = 0; t < 3; t++) {
    int ntile = wave + t * 8;
    v8f acc = {};
    const unsigned short* bbase = w2 + ntile * 16 * FDIM;
#pragma unroll
    for (int k0 = 0; k0 < FDIM; k0 += 32) {
      v16bf a = load_fragA(sH, FDIM, k0, nl, hh);
      v16bf b = load_fragB(bbase, FDIM, k0, nl, hh);
      acc = __builtin_amdgcn_wmma_f32_16x16x32_bf16(false, a, false, b,
                                                    (short)0, acc, false, false);
    }
    int n = ntile * 16 + nl;
    float bias2 = b2[n];
#pragma unroll
    for (int r = 0; r < 8; r++) {
      int m = r + hh * 8;
      sPW[m * F3 + n] = acc[r] + bias2;
    }
  }
  __syncthreads();

  // scatter s2 sums
  for (int i = tid; i < 16 * F3; i += THREADS) {
    int m = i / F3, n = i % F3;
    if (e0 + m < E)
      atomicAdd(&sum_s2[(long long)sDst[m] * F3 + n], sPW[m * F3 + n]);
  }
  // scatter uv sums (v_new[src])
  for (int i = tid; i < 16 * FDIM * 3; i += THREADS) {
    int m = i / (FDIM * 3), q = i % (FDIM * 3);
    if (e0 + m < E)
      atomicAdd(&uv_sum[(long long)sDst[m] * FDIM * 3 + q],
                v_new[(long long)sSrc[m] * FDIM * 3 + q]);
  }
}

// ---------------------------------------------------------------- finalize
__global__ void finalize_kernel(const float* __restrict__ s_new,
                                const float* __restrict__ v_new,
                                const float* __restrict__ sum_s2,
                                const float* __restrict__ uv_sum,
                                const float* __restrict__ deg,
                                float* out, int N) {
  long long NF = (long long)N * FDIM;
  long long i = (long long)blockIdx.x * blockDim.x + threadIdx.x;
  if (i >= NF) return;
  int n = (int)(i / FDIM), f = (int)(i % FDIM);
  float inv = 1.0f / fmaxf(deg[n], 1.0f);
  long long vb = i * 3;
  float u0 = uv_sum[vb + 0] * inv;
  float u1 = uv_sum[vb + 1] * inv;
  float u2 = uv_sum[vb + 2] * inv;
  float avv = sum_s2[(long long)n * F3 + f] * inv;
  float asv = sum_s2[(long long)n * F3 + FDIM + f] * inv;
  float ass = sum_s2[(long long)n * F3 + 2 * FDIM + f] * inv;
  float q = u0 * u0 + u1 * u1 + u2 * u2;
  float ds2 = q / (q + 1e-5f) * asv + ass;     // sum((uv/norm)^2)*asv + ass
  float* out_v = out;
  float* out_s = out + NF * 3;
  out_v[vb + 0] = v_new[vb + 0] + u0 * avv;
  out_v[vb + 1] = v_new[vb + 1] + u1 * avv;
  out_v[vb + 2] = v_new[vb + 2] + u2 * avv;
  out_s[i] = s_new[i] + ds2;
}

// ---------------------------------------------------------------- launch
extern "C" void kernel_launch(void* const* d_in, const int* in_sizes, int n_in,
                              void* d_out, int out_size, void* d_ws, size_t ws_size,
                              hipStream_t stream) {
  const float* x     = (const float*)d_in[0];
  const float* v     = (const float*)d_in[1];
  const float* s     = (const float*)d_in[2];
  const float* ms1_w = (const float*)d_in[3];
  const float* ms1_b = (const float*)d_in[4];
  const float* ms2_w = (const float*)d_in[5];
  const float* ms2_b = (const float*)d_in[6];
  const float* mv_w  = (const float*)d_in[7];
  const float* mv_b  = (const float*)d_in[8];
  const float* us1_w = (const float*)d_in[9];
  const float* us1_b = (const float*)d_in[10];
  const float* us2_w = (const float*)d_in[11];
  const float* us2_b = (const float*)d_in[12];
  const int*   src   = (const int*)d_in[13];
  const int*   dst   = (const int*)d_in[14];

  const int E = in_sizes[13];
  const int N = in_sizes[0] / 3;
  const size_t NF = (size_t)N * FDIM;

  float* ws     = (float*)d_ws;
  float* s_new  = ws;
  float* v_new  = s_new + NF;
  float* sum_s2 = v_new + NF * 3;
  float* uv_sum = sum_s2 + NF * 3;
  float* deg    = uv_sum + NF * 3;
  unsigned short* wbf = (unsigned short*)(deg + (((size_t)N + 3) & ~(size_t)3));
  unsigned short* w1b = wbf;
  unsigned short* w2b = w1b + FDIM * FDIM;
  unsigned short* u1b = w2b + F3 * FDIM;
  unsigned short* u2b = u1b + FDIM * FDIM;

  const int tiles = (E + 15) / 16;

  init_kernel<<<2048, THREADS, 0, stream>>>(v, s, v_new, s_new, sum_s2, uv_sum, deg, N);
  cvt_kernel<<<512, THREADS, 0, stream>>>(ms1_w, ms2_w, us1_w, us2_w, wbf);
  deg_kernel<<<(E + THREADS - 1) / THREADS, THREADS, 0, stream>>>(dst, deg, E);
  pass1_kernel<<<tiles, THREADS, 0, stream>>>(x, v, s, w1b, ms1_b, w2b, ms2_b,
                                              mv_w, mv_b, src, dst, s_new, v_new, E);
  pass2_kernel<<<tiles, THREADS, 0, stream>>>(u1b, us1_b, u2b, us2_b, src, dst,
                                              s_new, v_new, sum_s2, uv_sum, E);
  finalize_kernel<<<(int)((NF + THREADS - 1) / THREADS), THREADS, 0, stream>>>(
      s_new, v_new, sum_s2, uv_sum, deg, (float*)d_out, N);
}
